// Net_67259187855635
// MI455X (gfx1250) — matrix operand order repaired and checked by reading, not compiled, not run on hardware
//
#include <hip/hip_runtime.h>

typedef __attribute__((ext_vector_type(16))) _Float16 v16h;
typedef __attribute__((ext_vector_type(8)))  _Float16 v8h;
typedef __attribute__((ext_vector_type(8)))  float    v8f;

#define F_IN 256
#define HID  128
#define NCLS 47
#define NPAD 48

// ---------- weight conversion / transpose (run once per call, tiny) ----------
__global__ void k_w1t(const float* __restrict__ W1, _Float16* __restrict__ W1T) {
    int idx = blockIdx.x * 256 + threadIdx.x;       // 256*128 = 32768 threads
    int k = idx >> 7;                               // 0..255
    int n = idx & 127;                              // 0..127
    W1T[n * F_IN + k] = (_Float16)W1[k * HID + n];
}

__global__ void k_w2t(const float* __restrict__ W2, _Float16* __restrict__ W2T) {
    int idx = blockIdx.x * 256 + threadIdx.x;       // NPAD*HID = 6144 threads
    if (idx >= NPAD * HID) return;
    int n = idx >> 7;                               // 0..47 (padded col)
    int k = idx & 127;                              // 0..127
    W2T[n * HID + k] = (n < NCLS) ? (_Float16)W2[k * NCLS + n] : (_Float16)0.0f;
}

// ---------- degree / normalization ----------
__global__ void k_deg(const int* __restrict__ dst, float* __restrict__ deg, int E) {
    int e = blockIdx.x * 256 + threadIdx.x;
    if (e < E) unsafeAtomicAdd(&deg[dst[e]], 1.0f);
}

__global__ void k_dinv(float* __restrict__ dinv, int n) {
    int i = blockIdx.x * 256 + threadIdx.x;
    if (i < n) dinv[i] = rsqrtf(dinv[i] + 1.0f);    // +1 for self-loop
}

// ---------- GEMM1: xw = x @ W1   (M=100000, K=256, N=128), f16 WMMA ----------
// Block = 256 threads = 8 waves; block covers 16 rows; wave w covers cols [16w,16w+16).
__global__ __launch_bounds__(256) void k_gemm1(const float* __restrict__ x,
                                               const _Float16* __restrict__ W1T,
                                               float* __restrict__ xw, int nrows) {
    __shared__ _Float16 lA[16][F_IN];               // 8 KB
    const int r0 = blockIdx.x * 16;
    for (int t = threadIdx.x; t < 16 * F_IN; t += 256) {
        int r = t >> 8, c = t & 255;
        lA[r][c] = (_Float16)x[(size_t)(r0 + r) * F_IN + c];
    }
    if (r0 + 16 < nrows)                            // speculative prefetch of next M-tile
        __builtin_prefetch(&x[(size_t)(r0 + 16) * F_IN + threadIdx.x * 16], 0, 1);
    __syncthreads();

    const int wave = threadIdx.x >> 5;
    const int lane = threadIdx.x & 31;
    const int n0 = wave * 16;
    const int m  = lane & 15;                       // row within tile / col within B
    const int kb = (lane >> 4) * 8;                 // unused; A and B both use kl
    const int kl = (lane >> 4) * 16;                // B: K sub-block per lane half

    v8f acc = {};
    for (int k0 = 0; k0 < F_IN; k0 += 32) {
        v16h a;
        v8h* ap = (v8h*)&a;
        ap[0] = *(const v8h*)&lA[m][k0 + kl];       // frag elems 0..7  -> K = k0+kl+0..7
        ap[1] = *(const v8h*)&lA[m][k0 + kl + 8];   // frag elems 8..15 -> K = k0+kl+8..15
        v16h b = *(const v16h*)(W1T + (size_t)(n0 + m) * F_IN + k0 + kl);
        acc = __builtin_amdgcn_wmma_f32_16x16x32_f16(false, a, false, b,
                                                     (short)0, acc, false, false);
    }
    const int rbase = r0 + (lane >> 4) * 8;
    const int col = n0 + m;
    for (int j = 0; j < 8; ++j)
        xw[(size_t)(rbase + j) * HID + col] = acc[j];
}

// ---------- edge scatter 1: agg[d] += xw[s] * dinv[s]*dinv[d], wave per edge ----------
__global__ __launch_bounds__(256) void k_scatter1(const int* __restrict__ src,
                                                  const int* __restrict__ dst,
                                                  const float* __restrict__ dinv,
                                                  const float* __restrict__ xw,
                                                  float* __restrict__ agg, int E) {
    const int e = blockIdx.x * 8 + (threadIdx.x >> 5);
    if (e >= E) return;                             // wave-uniform
    const int lane = threadIdx.x & 31;
    const int s = src[e], d = dst[e];
    const float w = dinv[s] * dinv[d];
    const float4 v = *(const float4*)(xw + (size_t)s * HID + lane * 4);
    float* base = agg + (size_t)d * HID + lane * 4;
    unsafeAtomicAdd(base + 0, v.x * w);
    unsafeAtomicAdd(base + 1, v.y * w);
    unsafeAtomicAdd(base + 2, v.z * w);
    unsafeAtomicAdd(base + 3, v.w * w);
}

// ---------- layer-1 epilogue: h16 = f16(relu(agg + self + b1)) ----------
__global__ void k_h16(const float* __restrict__ agg, const float* __restrict__ xw,
                      const float* __restrict__ dinv, const float* __restrict__ b1,
                      _Float16* __restrict__ h16, int total) {
    int idx = blockIdx.x * 256 + threadIdx.x;
    if (idx >= total) return;
    int i = idx >> 7, f = idx & 127;
    float di = dinv[i];
    float v = agg[idx] + xw[idx] * di * di + b1[f];
    h16[idx] = (_Float16)fmaxf(v, 0.0f);
}

// ---------- GEMM2: xw2 = h @ W2   (M=100000, K=128, N=48 padded), f16 WMMA ----------
__global__ __launch_bounds__(256) void k_gemm2(const _Float16* __restrict__ h16,
                                               const _Float16* __restrict__ W2T,
                                               float* __restrict__ xw2) {
    __shared__ _Float16 lA[16][HID];                // 4 KB
    const int r0 = blockIdx.x * 16;
    for (int t = threadIdx.x; t < 16 * HID; t += 256) {
        int r = t >> 7, c = t & 127;
        lA[r][c] = h16[(size_t)(r0 + r) * HID + c];
    }
    __syncthreads();

    const int wave = threadIdx.x >> 5;
    if (wave < 3) {                                 // 3 n-tiles cover 48 cols
        const int lane = threadIdx.x & 31;
        const int n0 = wave * 16;
        const int m  = lane & 15;
        const int kb = (lane >> 4) * 8;
        const int kl = (lane >> 4) * 16;
        v8f acc = {};
        for (int k0 = 0; k0 < HID; k0 += 32) {
            v16h a;
            v8h* ap = (v8h*)&a;
            ap[0] = *(const v8h*)&lA[m][k0 + kl];
            ap[1] = *(const v8h*)&lA[m][k0 + kl + 8];
            v16h b = *(const v16h*)(W2T + (size_t)(n0 + m) * HID + k0 + kl);
            acc = __builtin_amdgcn_wmma_f32_16x16x32_f16(false, a, false, b,
                                                         (short)0, acc, false, false);
        }
        const int rbase = r0 + (lane >> 4) * 8;
        const int col = n0 + m;
        for (int j = 0; j < 8; ++j)
            xw2[(size_t)(rbase + j) * NPAD + col] = acc[j];
    }
}

// ---------- edge scatter 2: 47 features, wave per edge ----------
__global__ __launch_bounds__(256) void k_scatter2(const int* __restrict__ src,
                                                  const int* __restrict__ dst,
                                                  const float* __restrict__ dinv,
                                                  const float* __restrict__ xw2,
                                                  float* __restrict__ agg2, int E) {
    const int e = blockIdx.x * 8 + (threadIdx.x >> 5);
    if (e >= E) return;
    const int lane = threadIdx.x & 31;
    const int s = src[e], d = dst[e];
    const float w = dinv[s] * dinv[d];
    const size_t sb = (size_t)s * NPAD, db = (size_t)d * NPAD;
    unsafeAtomicAdd(agg2 + db + lane, xw2[sb + lane] * w);
    if (lane < NCLS - 32)
        unsafeAtomicAdd(agg2 + db + 32 + lane, xw2[sb + 32 + lane] * w);
}

// ---------- final: logits = agg2 + self + b2, log_softmax, wave per node ----------
__global__ __launch_bounds__(256) void k_final(const float* __restrict__ agg2,
                                               const float* __restrict__ xw2,
                                               const float* __restrict__ dinv,
                                               const float* __restrict__ b2,
                                               float* __restrict__ out, int n) {
    const int i = blockIdx.x * 8 + (threadIdx.x >> 5);
    if (i >= n) return;
    const int lane = threadIdx.x & 31;
    float d2 = dinv[i]; d2 *= d2;
    const size_t base = (size_t)i * NPAD;
    const bool has2 = lane < (NCLS - 32);
    const int f2 = 32 + lane;
    float v1 = agg2[base + lane] + xw2[base + lane] * d2 + b2[lane];
    float v2 = has2 ? (agg2[base + f2] + xw2[base + f2] * d2 + b2[f2]) : -3.402823e38f;
    float mx = fmaxf(v1, v2);
    for (int off = 16; off; off >>= 1) mx = fmaxf(mx, __shfl_xor(mx, off, 32));
    float sm = __expf(v1 - mx) + (has2 ? __expf(v2 - mx) : 0.0f);
    for (int off = 16; off; off >>= 1) sm += __shfl_xor(sm, off, 32);
    const float ls = __logf(sm);
    out[(size_t)i * NCLS + lane] = v1 - mx - ls;
    if (has2) out[(size_t)i * NCLS + f2] = v2 - mx - ls;
}

extern "C" void kernel_launch(void* const* d_in, const int* in_sizes, int n_in,
                              void* d_out, int out_size, void* d_ws, size_t ws_size,
                              hipStream_t stream) {
    const float* x  = (const float*)d_in[0];
    const int*   ei = (const int*)d_in[1];
    const float* W1 = (const float*)d_in[2];
    const float* b1 = (const float*)d_in[3];
    const float* W2 = (const float*)d_in[4];
    const float* b2 = (const float*)d_in[5];
    const int N = in_sizes[0] / F_IN;               // 100000 (multiple of 16)
    const int E = in_sizes[1] / 2;                  // 1600000
    const int* src = ei;
    const int* dst = ei + E;

    // ---- workspace layout (all offsets 4 KB aligned) ----
    char* ws = (char*)d_ws;
    float*    dinv = (float*)(ws + 0);                        //   400,000 B
    _Float16* W1T  = (_Float16*)(ws + 401408);                //    65,536 B
    _Float16* W2T  = (_Float16*)(ws + 466944);                //    12,288 B
    float*    xw   = (float*)(ws + 479232);                   // 51,200,000 B
    float*    agg1 = (float*)(ws + 51679232);                 // 51,200,000 B
    _Float16* h16  = (_Float16*)(ws + 102879232);             // 25,600,000 B
    float*    xw2  = xw;                                      // alias (xw dead after k_h16)
    float*    agg2 = agg1;                                    // alias (agg1 dead after k_h16)

    // weights -> f16, transposed for contiguous WMMA B-fragments
    k_w1t<<<(F_IN * HID) / 256, 256, 0, stream>>>(W1, W1T);
    k_w2t<<<(NPAD * HID + 255) / 256, 256, 0, stream>>>(W2, W2T);

    // degrees -> dinv
    hipMemsetAsync(dinv, 0, (size_t)N * 4, stream);
    k_deg<<<(E + 255) / 256, 256, 0, stream>>>(dst, dinv, E);
    k_dinv<<<(N + 255) / 256, 256, 0, stream>>>(dinv, N);

    // layer 1
    k_gemm1<<<N / 16, 256, 0, stream>>>(x, W1T, xw, N);
    hipMemsetAsync(agg1, 0, (size_t)N * HID * 4, stream);
    k_scatter1<<<(E + 7) / 8, 256, 0, stream>>>(src, dst, dinv, xw, agg1, E);
    k_h16<<<(N * HID + 255) / 256, 256, 0, stream>>>(agg1, xw, dinv, b1, h16, N * HID);

    // layer 2
    k_gemm2<<<N / 16, 256, 0, stream>>>(h16, W2T, xw2);
    hipMemsetAsync(agg2, 0, (size_t)N * NPAD * 4, stream);
    k_scatter2<<<(E + 7) / 8, 256, 0, stream>>>(src, dst, dinv, xw2, agg2, E);

    // epilogue: self term + bias + log_softmax
    k_final<<<(N + 7) / 8, 256, 0, stream>>>(agg2, xw2, dinv, b2, (float*)d_out, N);
}
